// DiffAttentionHead_4647154614312
// MI455X (gfx1250) — matrix operand order repaired and checked by reading, not compiled, not run on hardware
//
#include <hip/hip_runtime.h>
#include <hip/hip_bf16.h>

// Reference: B=16384, E=2048, H=16, HD=64.
// softmax over a singleton axis => A1=A2=1 identically; Q/K paths are dead.
// out = LN((1-lam) * (value @ W_v^T)) * gamma + beta, then @ W_o^T.
// GEMMs run as bf16x3 split-precision WMMA (fp32-class accuracy).

#define B_ROWS 16384
#define E_DIM  2048
#define HALF_E 1024

typedef unsigned short u16;
typedef __attribute__((ext_vector_type(16))) __bf16        v16bf;
typedef __attribute__((ext_vector_type(8)))  float         v8f;
typedef __attribute__((ext_vector_type(8)))  unsigned int  v8u;
typedef int v4i __attribute__((vector_size(16)));   // matches builtin param type

// gfx1250 async global->LDS path (ASYNCcnt), with sync fallback.
#if defined(__has_builtin)
#if __has_builtin(__builtin_amdgcn_global_load_async_to_lds_b128) && \
    __has_builtin(__builtin_amdgcn_s_wait_asynccnt)
#define USE_ASYNC_LDS 1
#endif
#endif

#define AS_GLOBAL __attribute__((address_space(1)))
#define AS_LDS    __attribute__((address_space(3)))

// ---------- fp32 -> bf16 hi/lo split (RNE) ----------
__device__ __forceinline__ u16 f32_to_bf16_rne(float x) {
    unsigned u = __float_as_uint(x);
    u += 0x7FFFu + ((u >> 16) & 1u);
    return (u16)(u >> 16);
}
__device__ __forceinline__ void bf16_split(float x, u16& h, u16& l) {
    h = f32_to_bf16_rne(x);
    float fh = __uint_as_float(((unsigned)h) << 16);
    l = f32_to_bf16_rne(x - fh);
}
__device__ __forceinline__ void cvt_pack2(float a, float b, unsigned& uh, unsigned& ul) {
    u16 ah, al, bh, bl;
    bf16_split(a, ah, al);
    bf16_split(b, bh, bl);
    uh = (unsigned)ah | ((unsigned)bh << 16);
    ul = (unsigned)al | ((unsigned)bl << 16);
}

// Build a v16bf fragment from two 16-byte LDS chunks.
__device__ __forceinline__ v16bf ld_frag(const u16* p0, const u16* p1) {
    const uint4 x = *(const uint4*)p0;
    const uint4 y = *(const uint4*)p1;
    v8u r;
    r[0] = x.x; r[1] = x.y; r[2] = x.z; r[3] = x.w;
    r[4] = y.x; r[5] = y.y; r[6] = y.z; r[7] = y.w;
    return __builtin_bit_cast(v16bf, r);
}

// ---------------------------------------------------------------------------
// One-time elementwise fp32 -> (bf16 hi, bf16 lo) split, 4 floats/thread.
// ---------------------------------------------------------------------------
__global__ __launch_bounds__(256)
void split_kernel(const float4* __restrict__ src, uint2* __restrict__ hi,
                  uint2* __restrict__ lo, int n4)
{
    const int i = blockIdx.x * 256 + threadIdx.x;
    if (i >= n4) return;
    const float4 v = src[i];
    unsigned uh0, ul0, uh1, ul1;
    cvt_pack2(v.x, v.y, uh0, ul0);
    cvt_pack2(v.z, v.w, uh1, ul1);
    uint2 h; h.x = uh0; h.y = uh1;
    uint2 l; l.x = ul0; l.y = ul1;
    hi[i] = h;
    lo[i] = l;
}

// ---------------------------------------------------------------------------
// GEMM: C[M][N] = A[M][K] * B[N][K]^T with pre-split bf16 hi/lo operands.
// Block: 256 threads = 8 waves; block tile 128x128; BK=32.
// Wave (2x4): 64x32 output = 4x2 WMMA tiles -> 24 v_wmma per K-step.
// Staging: GLOBAL_LOAD_ASYNC_TO_LDS_B128 double-buffered (ASYNCcnt) so the
// DMA for slab k+1 overlaps the WMMA work on slab k. Sync fallback included.
// ---------------------------------------------------------------------------
template<int K>
__global__ __launch_bounds__(256)
void gemm_bf16x3(const u16* __restrict__ Ahi, const u16* __restrict__ Alo,
                 const u16* __restrict__ Bhi, const u16* __restrict__ Blo,
                 float* __restrict__ C, int N)
{
#ifdef USE_ASYNC_LDS
    constexpr int NBUF = 2;
#else
    constexpr int NBUF = 1;
#endif
    __shared__ alignas(16) u16 sAhi[NBUF][128 * 32];
    __shared__ alignas(16) u16 sAlo[NBUF][128 * 32];
    __shared__ alignas(16) u16 sBhi[NBUF][128 * 32];
    __shared__ alignas(16) u16 sBlo[NBUF][128 * 32];

    const int tid   = threadIdx.x;
    const int lane  = tid & 31;
    const int wid   = tid >> 5;
    const int waveM = (wid >> 2) * 64;   // 0 or 64
    const int waveN = (wid & 3) * 32;    // 0,32,64,96
    const size_t blockM = (size_t)blockIdx.y * 128;
    const size_t blockN = (size_t)blockIdx.x * 128;

    // Staging: 2 threads per row, 8 consecutive u16 (16B) each.
    const int sr   = tid >> 1;           // 0..127
    const int sh   = (tid & 1) * 16;     // 0 or 16
    const int soff = sr * 32 + sh;

    const u16* gAh = Ahi + (blockM + sr) * (size_t)K + sh;
    const u16* gAl = Alo + (blockM + sr) * (size_t)K + sh;
    const u16* gBh = Bhi + (blockN + sr) * (size_t)K + sh;
    const u16* gBl = Blo + (blockN + sr) * (size_t)K + sh;

    v8f acc[4][2];
    #pragma unroll
    for (int i = 0; i < 4; ++i)
        #pragma unroll
        for (int j = 0; j < 2; ++j)
            #pragma unroll
            for (int e = 0; e < 8; ++e) acc[i][j][e] = 0.0f;

    // Fragment addressing (ISA 7.12.2 16-bit layouts, wave32):
    const int ca = (lane >> 4) * 8;   // A: lane<16 -> K{0..7,16..23}; else +8
    const int cb = (lane >> 4) * 8;   // B: lane<16 -> K{0..7,16..23}; else +8
    const int ml = lane & 15;

    auto compute = [&](int buf) {
        v16bf aH[4], aL[4];
        #pragma unroll
        for (int mt = 0; mt < 4; ++mt) {
            const int m = waveM + mt * 16 + ml;
            aH[mt] = ld_frag(sAhi[buf] + m * 32 + ca, sAhi[buf] + m * 32 + ca + 16);
            aL[mt] = ld_frag(sAlo[buf] + m * 32 + ca, sAlo[buf] + m * 32 + ca + 16);
        }
        v16bf bH[2], bL[2];
        #pragma unroll
        for (int nt = 0; nt < 2; ++nt) {
            const int n = waveN + nt * 16 + ml;
            bH[nt] = ld_frag(sBhi[buf] + n * 32 + cb, sBhi[buf] + n * 32 + cb + 16);
            bL[nt] = ld_frag(sBlo[buf] + n * 32 + cb, sBlo[buf] + n * 32 + cb + 16);
        }
        // bf16x3: hi*hi + hi*lo + lo*hi (lo*lo ~ 2^-18, dropped)
        #pragma unroll
        for (int mt = 0; mt < 4; ++mt) {
            #pragma unroll
            for (int nt = 0; nt < 2; ++nt) {
                acc[mt][nt] = __builtin_amdgcn_wmma_f32_16x16x32_bf16(
                    false, aH[mt], false, bH[nt], (short)0, acc[mt][nt], false, false);
                acc[mt][nt] = __builtin_amdgcn_wmma_f32_16x16x32_bf16(
                    false, aH[mt], false, bL[nt], (short)0, acc[mt][nt], false, false);
                acc[mt][nt] = __builtin_amdgcn_wmma_f32_16x16x32_bf16(
                    false, aL[mt], false, bH[nt], (short)0, acc[mt][nt], false, false);
            }
        }
    };

#ifdef USE_ASYNC_LDS
    // Async global->LDS DMA staging, no VGPR transit.
    auto issue = [&](int buf, int k0) {
        __builtin_amdgcn_global_load_async_to_lds_b128(
            (AS_GLOBAL v4i*)(gAh + k0), (AS_LDS v4i*)(&sAhi[buf][soff]), 0, 0);
        __builtin_amdgcn_global_load_async_to_lds_b128(
            (AS_GLOBAL v4i*)(gAl + k0), (AS_LDS v4i*)(&sAlo[buf][soff]), 0, 0);
        __builtin_amdgcn_global_load_async_to_lds_b128(
            (AS_GLOBAL v4i*)(gBh + k0), (AS_LDS v4i*)(&sBhi[buf][soff]), 0, 0);
        __builtin_amdgcn_global_load_async_to_lds_b128(
            (AS_GLOBAL v4i*)(gBl + k0), (AS_LDS v4i*)(&sBlo[buf][soff]), 0, 0);
    };

    issue(0, 0);
    __builtin_amdgcn_s_wait_asynccnt(0);
    __syncthreads();
    int cur = 0;
    for (int k0 = 0; k0 < K; k0 += 32) {
        if (k0 + 32 < K) issue(cur ^ 1, k0 + 32);   // DMA overlaps WMMA below
        compute(cur);
        __builtin_amdgcn_s_wait_asynccnt(0);        // next slab landed in LDS
        __syncthreads();
        cur ^= 1;
    }
#else
    uint4* dAh = (uint4*)(sAhi[0] + soff);
    uint4* dAl = (uint4*)(sAlo[0] + soff);
    uint4* dBh = (uint4*)(sBhi[0] + soff);
    uint4* dBl = (uint4*)(sBlo[0] + soff);
    for (int k0 = 0; k0 < K; k0 += 32) {
        *dAh = *(const uint4*)(gAh + k0);
        *dAl = *(const uint4*)(gAl + k0);
        *dBh = *(const uint4*)(gBh + k0);
        *dBl = *(const uint4*)(gBl + k0);
        if (k0 + 32 < K) {
            __builtin_prefetch(gAh + k0 + 32, 0, 1);
            __builtin_prefetch(gAl + k0 + 32, 0, 1);
            __builtin_prefetch(gBh + k0 + 32, 0, 1);
            __builtin_prefetch(gBl + k0 + 32, 0, 1);
        }
        __syncthreads();
        compute(0);
        __syncthreads();
    }
#endif

    // ---- store: C/D layout: VGPR r -> M = r + 8*(lane>=16); N = lane%16 ----
    const int mbase = (lane >> 4) << 3;
    const int ncol  = lane & 15;
    #pragma unroll
    for (int mt = 0; mt < 4; ++mt) {
        #pragma unroll
        for (int nt = 0; nt < 2; ++nt) {
            const size_t n = blockN + waveN + nt * 16 + ncol;
            #pragma unroll
            for (int r = 0; r < 8; ++r) {
                const size_t m = blockM + waveM + mt * 16 + mbase + r;
                C[m * (size_t)N + n] = acc[mt][nt][r];
            }
        }
    }
}

// ---------------------------------------------------------------------------
// lam = exp(sum lq1*lk1) - exp(sum lq2*lk2) + 0.8 ; store c = 1 - lam
// ---------------------------------------------------------------------------
__global__ void lambda_kernel(const float* __restrict__ lq1, const float* __restrict__ lk1,
                              const float* __restrict__ lq2, const float* __restrict__ lk2,
                              float* __restrict__ c)
{
    const int t = threadIdx.x;  // 32 threads, HD=64 -> 2 elements/lane
    float s1 = lq1[t] * lk1[t] + lq1[t + 32] * lk1[t + 32];
    float s2 = lq2[t] * lk2[t] + lq2[t + 32] * lk2[t + 32];
    #pragma unroll
    for (int off = 16; off > 0; off >>= 1) {
        s1 += __shfl_xor(s1, off);
        s2 += __shfl_xor(s2, off);
    }
    if (t == 0) {
        const float lam = expf(s1) - expf(s2) + 0.8f;
        c[0] = 1.0f - lam;
    }
}

// ---------------------------------------------------------------------------
// Rowwise LayerNorm of c*V over 1024 cols (stats in fp32); writes the bf16
// hi/lo split of the normalized rows directly (gemm2 consumes split operands).
// One wave per row.
// ---------------------------------------------------------------------------
__global__ __launch_bounds__(256)
void ln_split_kernel(const float* __restrict__ V, const float* __restrict__ gamma,
                     const float* __restrict__ beta, const float* __restrict__ csp,
                     uint2* __restrict__ Xhi, uint2* __restrict__ Xlo)
{
    const int lane = threadIdx.x & 31;
    const int wid  = threadIdx.x >> 5;
    const size_t row = (size_t)blockIdx.x * 8 + wid;
    const float4* p = (const float4*)(V + row * HALF_E);
    const float cs = csp[0];

    float4 v[8];
    float s = 0.f, sq = 0.f;
    #pragma unroll
    for (int j = 0; j < 8; ++j) {
        float4 t = p[j * 32 + lane];
        t.x *= cs; t.y *= cs; t.z *= cs; t.w *= cs;
        v[j] = t;
        s  += t.x + t.y + t.z + t.w;
        sq += t.x * t.x + t.y * t.y + t.z * t.z + t.w * t.w;
    }
    #pragma unroll
    for (int off = 16; off > 0; off >>= 1) {
        s  += __shfl_xor(s, off);
        sq += __shfl_xor(sq, off);
    }
    const float mu  = s * (1.0f / HALF_E);
    const float var = sq * (1.0f / HALF_E) - mu * mu;
    const float rs  = rsqrtf(var + 1e-5f);
    #pragma unroll
    for (int j = 0; j < 8; ++j) {
        const float4 g = ((const float4*)gamma)[j * 32 + lane];
        const float4 b = ((const float4*)beta )[j * 32 + lane];
        float4 t = v[j];
        t.x = (t.x - mu) * rs * g.x + b.x;
        t.y = (t.y - mu) * rs * g.y + b.y;
        t.z = (t.z - mu) * rs * g.z + b.z;
        t.w = (t.w - mu) * rs * g.w + b.w;
        unsigned uh0, ul0, uh1, ul1;
        cvt_pack2(t.x, t.y, uh0, ul0);
        cvt_pack2(t.z, t.w, uh1, ul1);
        uint2 h; h.x = uh0; h.y = uh1;
        uint2 l; l.x = ul0; l.y = ul1;
        const size_t idx = row * 256 + (size_t)(j * 32 + lane);
        Xhi[idx] = h;
        Xlo[idx] = l;
    }
}

// ---------------------------------------------------------------------------
extern "C" void kernel_launch(void* const* d_in, const int* in_sizes, int n_in,
                              void* d_out, int out_size, void* d_ws, size_t ws_size,
                              hipStream_t stream)
{
    // Inputs: query, key, value, W_q, W_k, W_v, W_o,
    //         lambda_q1, lambda_k1, lambda_q2, lambda_k2, ln_gamma, ln_beta
    const float* value = (const float*)d_in[2];
    const float* W_v   = (const float*)d_in[5];
    const float* W_o   = (const float*)d_in[6];
    const float* lq1   = (const float*)d_in[7];
    const float* lk1   = (const float*)d_in[8];
    const float* lq2   = (const float*)d_in[9];
    const float* lk2   = (const float*)d_in[10];
    const float* gamma = (const float*)d_in[11];
    const float* beta  = (const float*)d_in[12];
    float* out = (float*)d_out;

    // ---- workspace layout (bytes) ----
    // [0,256):   scalar c
    // V fp32:    16384 x 1024 (64 MB)
    // Wv hi/lo:  1024 x 2048 u16 each; Wo hi/lo: 2048 x 1024 u16 each (16 MB)
    // Region R:  value hi/lo (67 MB each), later aliased by X hi/lo
    char* wsb = (char*)d_ws;
    float* cscale = (float*)wsb;
    float* V      = (float*)(wsb + 256);
    const size_t nV  = (size_t)B_ROWS * HALF_E;       // 16.7M
    const size_t nW  = (size_t)HALF_E * E_DIM;        // 2.1M (both weights)
    const size_t nA  = (size_t)B_ROWS * E_DIM;        // 33.5M
    u16* Wvh = (u16*)(wsb + 256 + nV * 4);
    u16* Wvl = Wvh + nW;
    u16* Woh = Wvl + nW;
    u16* Wol = Woh + nW;
    u16* Avh = Wol + nW;          // region R
    u16* Avl = Avh + nA;
    u16* Xh  = Avh;               // alias: value split dead after gemm1
    u16* Xl  = Xh + nV;

    lambda_kernel<<<1, 32, 0, stream>>>(lq1, lk1, lq2, lk2, cscale);

    // One-time bf16 hi/lo splits.
    split_kernel<<<(int)(nA / 4 / 256), 256, 0, stream>>>(
        (const float4*)value, (uint2*)Avh, (uint2*)Avl, (int)(nA / 4));
    split_kernel<<<(int)(nW / 4 / 256), 256, 0, stream>>>(
        (const float4*)W_v, (uint2*)Wvh, (uint2*)Wvl, (int)(nW / 4));
    split_kernel<<<(int)(nW / 4 / 256), 256, 0, stream>>>(
        (const float4*)W_o, (uint2*)Woh, (uint2*)Wol, (int)(nW / 4));

    // V = value @ W_v^T   (M=16384, N=1024, K=2048)
    gemm_bf16x3<E_DIM><<<dim3(HALF_E / 128, B_ROWS / 128), 256, 0, stream>>>(
        Avh, Avl, Wvh, Wvl, V, HALF_E);

    // X = LN(c*V)*gamma + beta, emitted as bf16 hi/lo
    ln_split_kernel<<<B_ROWS / 8, 256, 0, stream>>>(V, gamma, beta, cscale,
                                                    (uint2*)Xh, (uint2*)Xl);

    // out = X @ W_o^T     (M=16384, N=2048, K=1024)
    gemm_bf16x3<HALF_E><<<dim3(E_DIM / 128, B_ROWS / 128), 256, 0, stream>>>(
        Xh, Xl, Woh, Wol, out, E_DIM);
}
